// LeakyRNN_48533130444809
// MI455X (gfx1250) — matrix operand order repaired and checked by reading, not compiled, not run on hardware
//
#include <hip/hip_runtime.h>

#define NI 128
#define NH 512
#define NO 64
#define BB 64
#define TT 1024
#define ALPHAF 0.1f

typedef __attribute__((ext_vector_type(16))) __bf16 bf16x16;
typedef __attribute__((ext_vector_type(8)))  float  f32x8;
typedef __attribute__((ext_vector_type(4)))  unsigned u32x4;
typedef __attribute__((ext_vector_type(8)))  unsigned u32x8;

union ABFrag { bf16x16 v; unsigned u[8]; };

// round-to-nearest-even f32 -> bf16, two packed into one u32 (lo in [15:0])
__device__ __forceinline__ unsigned pack2_bf16(float lo, float hi) {
    unsigned a = __builtin_bit_cast(unsigned, lo);
    unsigned b = __builtin_bit_cast(unsigned, hi);
    a += 0x7FFFu + ((a >> 16) & 1u);
    b += 0x7FFFu + ((b >> 16) & 1u);
    return (a >> 16) | (b & 0xFFFF0000u);
}

__device__ __forceinline__ unsigned short f2bf(float f) {
    unsigned a = __builtin_bit_cast(unsigned, f);
    a += 0x7FFFu + ((a >> 16) & 1u);
    return (unsigned short)(a >> 16);
}

// A fragment (16x32 bf16, rows mbase..mbase+15, K = ks*32..+31) from LDS
// stored as u32-packed bf16 pairs, row stride in u32 units.
// ISA layout: lane L holds row M = L%16; half = L/16 selects K octet;
// VGPR v: K = (v>=4?16:0) + half*8 + 2*(v%4)  (pairs contiguous in K).
__device__ __forceinline__ bf16x16 frag_a(const unsigned* sh, int stride,
                                          int mbase, int ks, int lane) {
    ABFrag f;
    const unsigned* row = sh + (mbase + (lane & 15)) * stride
                             + ks * 16 + ((lane >> 4) << 2);
    #pragma unroll
    for (int v = 0; v < 8; ++v)
        f.u[v] = row[((v & 4) << 1) + (v & 3)];
    return f.v;
}

// B fragment (32x16 bf16): element (K, N) = W[nbase+N][K] with W rows in LDS.
// ISA layout: lane L holds col N = L%16; half = L/16 selects K octet;
// VGPR v: K = (v>=4?16:0) + half*8 + 2*(v%4)  (pairs contiguous in K).
__device__ __forceinline__ bf16x16 frag_b(const unsigned* sh, int stride,
                                          int nbase, int ks, int lane) {
    ABFrag f;
    const unsigned* row = sh + (nbase + (lane & 15)) * stride
                             + ks * 16 + ((lane >> 4) << 2);
    #pragma unroll
    for (int v = 0; v < 8; ++v)
        f.u[v] = row[((v & 4) << 1) + (v & 3)];
    return f.v;
}

#define WMMA_BF16(a, b, c) \
    __builtin_amdgcn_wmma_f32_16x16x32_bf16(false, (a), false, (b), (short)0, (c), false, false)

// ---------------------------------------------------------------------------
// Kernel 1: ui = u @ Wi^T, written into the As output region.
// Grid: (BT/128) x (NH/64); each WG computes a 128x64 block. 8 waves, 4 tiles each.
// ---------------------------------------------------------------------------
__global__ __launch_bounds__(256) void uiproj_kernel(
    const float* __restrict__ u, const float* __restrict__ Wi,
    float* __restrict__ As)
{
    __shared__ unsigned shU[128 * 68];   // 128 rows x 64 K-pairs, padded stride 68
    __shared__ unsigned shW[64 * 68];
    const int tid = threadIdx.x, lane = tid & 31, wave = tid >> 5;
    const int row0 = blockIdx.x * 128;
    const int h0   = blockIdx.y * 64;

    for (int i = tid; i < 128 * 64; i += 256) {
        int r = i >> 6, kp = i & 63;
        const float* s = u + (size_t)(row0 + r) * NI + kp * 2;
        shU[r * 68 + kp] = pack2_bf16(s[0], s[1]);
    }
    for (int i = tid; i < 64 * 64; i += 256) {
        int r = i >> 6, kp = i & 63;
        const float* s = Wi + (size_t)(h0 + r) * NI + kp * 2;
        shW[r * 68 + kp] = pack2_bf16(s[0], s[1]);
    }
    __syncthreads();

    const int Mt = (wave >> 1) * 32;     // two 16-row tiles
    const int Nt = (wave & 1) * 32;      // two 16-col tiles
    f32x8 acc00 = {}, acc01 = {}, acc10 = {}, acc11 = {};
    #pragma unroll
    for (int ks = 0; ks < NI / 32; ++ks) {
        bf16x16 a0 = frag_a(shU, 68, Mt,      ks, lane);
        bf16x16 a1 = frag_a(shU, 68, Mt + 16, ks, lane);
        bf16x16 b0 = frag_b(shW, 68, Nt,      ks, lane);
        bf16x16 b1 = frag_b(shW, 68, Nt + 16, ks, lane);
        acc00 = WMMA_BF16(a0, b0, acc00);
        acc01 = WMMA_BF16(a0, b1, acc01);
        acc10 = WMMA_BF16(a1, b0, acc10);
        acc11 = WMMA_BF16(a1, b1, acc11);
    }
    const int n = lane & 15, halfm = (lane >> 4) << 3;
    #pragma unroll
    for (int r = 0; r < 8; ++r) {
        size_t rm0 = (size_t)(row0 + Mt +      halfm + r) * NH + h0;
        size_t rm1 = (size_t)(row0 + Mt + 16 + halfm + r) * NH + h0;
        As[rm0 + Nt + n]      = acc00[r];
        As[rm0 + Nt + 16 + n] = acc01[r];
        As[rm1 + Nt + n]      = acc10[r];
        As[rm1 + Nt + 16 + n] = acc11[r];
    }
}

// ---------------------------------------------------------------------------
// Kernel 2: persistent leaky-RNN scan. 8 WGs; WG w owns hidden cols [64w,64w+64).
// Wr block + full bf16 state in LDS; f32 master state for owned cols in regs.
// One device-wide atomic barrier per step; state exchanged via ping-pong bf16
// global buffer, pulled back into LDS by the Tensor Data Mover (one
// tensor_load_to_lds per WG per step, with native LDS row padding).
// ---------------------------------------------------------------------------
__global__ __launch_bounds__(256) void scan_kernel(
    const float* __restrict__ Wr,
    float* __restrict__ X, float* __restrict__ As, float* __restrict__ Phis,
    unsigned short* __restrict__ xg,   // [2][BB*NH] bf16 ping-pong state
    unsigned* __restrict__ cnt)        // grid barrier counter (pre-zeroed)
{
    __shared__ unsigned shW[64 * 260];   // 64 Wr rows x 256 K-pairs, stride 260
    __shared__ unsigned shX[64 * 260];   // full state, bf16 pairs, stride 260
    const int tid = threadIdx.x, lane = tid & 31, wave = tid >> 5;
    const int hbase = blockIdx.x * 64;

    for (int i = tid; i < 64 * 256; i += 256) {
        int r = i >> 8, kp = i & 255;
        const float* s = Wr + (size_t)(hbase + r) * NH + kp * 2;
        shW[r * 260 + kp] = pack2_bf16(s[0], s[1]);
    }
    for (int i = tid; i < 64 * 260; i += 256) shX[i] = 0u;   // x0 = 0
    __syncthreads();

    // --- TDM descriptor, constant group 1 (ISA 8.4) ---
    // 2D copy: 64 rows x 256 DWORDs; DWORD elements (data_size=2);
    // pad_enable, pad_interval=7 (after every 256 DW), pad_amount=3 (4 DW)
    // -> reproduces the 260-DWORD padded LDS row stride.
    u32x8 g1;
    g1[0] = (2u << 16) | (1u << 20) | (7u << 22) | (3u << 25);
    g1[1] = (256u & 0xFFFFu) << 16;          // tensor_dim0[15:0] = 256
    g1[2] = (256u >> 16) | (64u << 16);      // tensor_dim0[31:16] | tensor_dim1[15:0]=64
    g1[3] = (64u >> 16) | (256u << 16);      // tensor_dim1[31:16] | tile_dim0=256
    g1[4] = 64u;                             // tile_dim1=64, tile_dim2=0
    g1[5] = 256u;                            // tensor_dim0_stride[31:0] = 256
    g1[6] = (16384u & 0xFFFFu) << 16;        // stride0[47:32]=0 | stride1[15:0]
    g1[7] = 16384u >> 16;                    // stride1[47:16]
    const unsigned shx_lds = (unsigned)(uintptr_t)&shX[0];

    const int Mt  = (wave >> 1) * 16;
    const int Nt0 = (wave & 1) * 32;
    const int Nt1 = Nt0 + 16;
    const int n = lane & 15, halfm = (lane >> 4) << 3;

    f32x8 x0 = {}, x1 = {};              // f32 master state, D-tile layout

    for (int t = 0; t < TT; ++t) {
        f32x8 acc0 = {}, acc1 = {};
        #pragma unroll 4
        for (int ks = 0; ks < NH / 32; ++ks) {
            bf16x16 a  = frag_a(shX, 260, Mt,  ks, lane);
            bf16x16 b0 = frag_b(shW, 260, Nt0, ks, lane);
            bf16x16 b1 = frag_b(shW, 260, Nt1, ks, lane);
            acc0 = WMMA_BF16(a, b0, acc0);
            acc1 = WMMA_BF16(a, b1, acc1);
        }
        unsigned short* xw = xg + (size_t)(t & 1) * (BB * NH);
        #pragma unroll
        for (int r = 0; r < 8; ++r) {
            int brow = Mt + halfm + r;
            size_t base = ((size_t)brow * TT + t) * NH + hbase;
            size_t i0 = base + Nt0 + n;
            size_t i1 = base + Nt1 + n;
            float a0 = acc0[r] + As[i0];          // As currently holds ui
            float a1 = acc1[r] + As[i1];
            __builtin_prefetch(&As[i0 + NH], 0, 3);   // next timestep's ui
            __builtin_prefetch(&As[i1 + NH], 0, 3);
            float p0 = tanhf(a0), p1 = tanhf(a1);
            float xn0 = ALPHAF * p0 + (1.0f - ALPHAF) * x0[r];
            float xn1 = ALPHAF * p1 + (1.0f - ALPHAF) * x1[r];
            As[i0] = a0;   As[i1] = a1;
            Phis[i0] = p0; Phis[i1] = p1;
            X[i0] = xn0;   X[i1] = xn1;
            x0[r] = xn0;   x1[r] = xn1;
            xw[brow * NH + hbase + Nt0 + n] = f2bf(xn0);
            xw[brow * NH + hbase + Nt1 + n] = f2bf(xn1);
        }
        // ---- device-wide barrier (8 WGs) ----
        __threadfence();
        __syncthreads();
        if (tid == 0) {
            atomicAdd(cnt, 1u);
            unsigned target = 8u * (unsigned)(t + 1);
            while (__hip_atomic_load(cnt, __ATOMIC_ACQUIRE,
                                     __HIP_MEMORY_SCOPE_AGENT) < target)
                __builtin_amdgcn_s_sleep(1);
        }
        __syncthreads();
        if (t + 1 < TT) {
            // reload full bf16 state into LDS via the Tensor Data Mover
            if (wave == 0) {
                unsigned long long ga = (unsigned long long)(uintptr_t)xw;
                u32x4 g0;
                g0[0] = 1u;                                   // count=1
                g0[1] = shx_lds;                              // lds_addr
                g0[2] = (unsigned)(ga & 0xFFFFFFFFu);         // global_addr[31:0]
                g0[3] = (unsigned)((ga >> 32) & 0x1FFFFFFu)   // global_addr[56:32]
                        | (2u << 30);                         // type=2 (image)
                asm volatile("tensor_load_to_lds %0, %1"
                             :: "s"(g0), "s"(g1) : "memory");
                __builtin_amdgcn_s_wait_tensorcnt(0);
            }
            __syncthreads();
        }
    }
}

// ---------------------------------------------------------------------------
// Kernel 3: y_hat = X[:, T-1, :] @ Wo^T  (64x512 @ 512x64). One WG.
// ---------------------------------------------------------------------------
__global__ __launch_bounds__(256) void head_kernel(
    const float* __restrict__ X, const float* __restrict__ Wo,
    float* __restrict__ yhat)
{
    __shared__ unsigned shXl[64 * 260];
    __shared__ unsigned shW[64 * 260];
    const int tid = threadIdx.x, lane = tid & 31, wave = tid >> 5;
    for (int i = tid; i < 64 * 256; i += 256) {
        int r = i >> 8, kp = i & 255;
        const float* s = X + ((size_t)r * TT + (TT - 1)) * NH + kp * 2;
        shXl[r * 260 + kp] = pack2_bf16(s[0], s[1]);
        const float* w = Wo + (size_t)r * NH + kp * 2;
        shW[r * 260 + kp] = pack2_bf16(w[0], w[1]);
    }
    __syncthreads();
    const int Mt  = (wave >> 1) * 16;
    const int Nt0 = (wave & 1) * 32, Nt1 = Nt0 + 16;
    f32x8 acc0 = {}, acc1 = {};
    #pragma unroll 4
    for (int ks = 0; ks < NH / 32; ++ks) {
        bf16x16 a  = frag_a(shXl, 260, Mt,  ks, lane);
        bf16x16 b0 = frag_b(shW, 260, Nt0, ks, lane);
        bf16x16 b1 = frag_b(shW, 260, Nt1, ks, lane);
        acc0 = WMMA_BF16(a, b0, acc0);
        acc1 = WMMA_BF16(a, b1, acc1);
    }
    const int n = lane & 15, halfm = (lane >> 4) << 3;
    #pragma unroll
    for (int r = 0; r < 8; ++r) {
        int b = Mt + halfm + r;
        yhat[(size_t)b * NO + Nt0 + n] = acc0[r];
        yhat[(size_t)b * NO + Nt1 + n] = acc1[r];
    }
}

// ---------------------------------------------------------------------------
extern "C" void kernel_launch(void* const* d_in, const int* in_sizes, int n_in,
                              void* d_out, int out_size, void* d_ws, size_t ws_size,
                              hipStream_t stream) {
    const float* u  = (const float*)d_in[0];
    const float* Wr = (const float*)d_in[1];
    const float* Wi = (const float*)d_in[2];
    const float* Wo = (const float*)d_in[3];
    const size_t BTN = (size_t)BB * TT * NH;
    float* X    = (float*)d_out;
    float* As   = X + BTN;
    float* Phis = As + BTN;
    float* yhat = Phis + BTN;

    unsigned short* xg = (unsigned short*)d_ws;                       // 2*64*512 bf16
    unsigned* cnt = (unsigned*)((char*)d_ws + 2u * BB * NH * sizeof(unsigned short));
    hipMemsetAsync(cnt, 0, 64, stream);   // barrier counter only; xg never read-before-write

    dim3 gA(BB * TT / 128, NH / 64);
    uiproj_kernel<<<gA, dim3(256), 0, stream>>>(u, Wi, As);
    scan_kernel<<<dim3(NH / 64), dim3(256), 0, stream>>>(Wr, X, As, Phis, xg, cnt);
    head_kernel<<<dim3(1), dim3(256), 0, stream>>>(X, Wo, yhat);
}
